// WindowAttention_80874234183835
// MI455X (gfx1250) — matrix operand order repaired and checked by reading, not compiled, not run on hardware
//
#include <hip/hip_runtime.h>
#include <math.h>

// ---------------------------------------------------------------------------
// Window attention, fused per-window kernel for gfx1250 (MI455X).
// B=8, N=4096, C=512, H=8, head_dim=64, window W=64, SHIFT by 32.
// One workgroup (512 threads = 16 wave32) handles one window end-to-end:
//   x(win) -> LDS(bf16) -> QKV gemm (WMMA bf16) -> per-head softmax(QK^T)
//   -> PV (WMMA) -> proj gemm (WMMA) + bias -> fp32 out (inverse roll).
// Intermediates never touch HBM. Weights are bf16-packed ONCE into d_ws in
// exact WMMA B-fragment order so every global B load is one contiguous
// 1KB/wave burst. LDS rows padded to kill bank conflicts.
// ---------------------------------------------------------------------------

typedef __attribute__((ext_vector_type(16))) __bf16 v16bf;
typedef __attribute__((ext_vector_type(8)))  float  v8f;
typedef unsigned short ushort_t;

#define C_DIM   512
#define QKV_DIM 1536
#define WTOK    64
#define NHEAD   8
#define HDIM    64
#define SEQ     4096
#define NWINB   64      // windows per batch
#define SCALE_Q 0.125f
#define LDSP    520     // padded row stride for [64][512] buffers (1040B % 256 = 16)
#define PT      72      // padded row stride for 64-wide buffers (144B % 256 = 144)

// LDS region sizes (elements)
#define R0_ELEMS 36864  // xs (64*520=33280) / ps (8*64*72=36864) -> max
#define R1_ELEMS 33280  // qs / os
#define R2_ELEMS 33280  // ks
#define R3_ELEMS 36864  // vst [8][64][72]

// round-to-nearest-even fp32 -> bf16
__device__ __forceinline__ ushort_t f2bf(float f) {
    unsigned int u = __float_as_uint(f);
    return (ushort_t)((u + 0x7FFFu + ((u >> 16) & 1u)) >> 16);
}

// A-fragment (16x32 bf16): lane holds k = 16*(lane>>4) + e : 16 contiguous.
// p points at row base + 16*(lane>>4); chunks at p, p+8.
__device__ __forceinline__ v16bf frag_a(const ushort_t* p) {
    v16bf r;
    ((uint4*)&r)[0] = *(const uint4*)(p);
    ((uint4*)&r)[1] = *(const uint4*)(p + 8);
    return r;
}
// B-fragment (32x16 bf16): lane holds k = 16*(lane>>4) + e : 16 contiguous.
__device__ __forceinline__ v16bf frag_b16(const ushort_t* p) {
    v16bf r;
    ((uint4*)&r)[0] = ((const uint4*)p)[0];
    ((uint4*)&r)[1] = ((const uint4*)p)[1];
    return r;
}

__device__ __forceinline__ v8f wmma_bf16(v16bf a, v16bf b, v8f c) {
    return __builtin_amdgcn_wmma_f32_16x16x32_bf16(false, a, false, b,
                                                   (short)0, c, false, false);
}

// ---------------- weight pre-pack: fp32 -> bf16 in B-fragment order ---------
// Packed layout: frag slot i = (ct*16 + kt)*32 + lane ; 16 bf16 per slot.
//   c = ct*16 + (lane&15);  k = kt*32 + 16*(lane>>4) + e
// A wave's B-fragment load becomes 32 lanes x 32B contiguous = 1KB burst.
#define NQ_SLOTS (96 * 16 * 32)   // 1536/16 col tiles * 512/32 k tiles * 32 lanes
#define NP_SLOTS (32 * 16 * 32)   // 512/16  col tiles * 512/32 k tiles * 32 lanes

__global__ void pack_weights_kernel(const float* __restrict__ wq,
                                    const float* __restrict__ wp,
                                    ushort_t* __restrict__ wqb,
                                    ushort_t* __restrict__ wpb) {
    int tid = blockIdx.x * blockDim.x + threadIdx.x;
    if (tid >= NQ_SLOTS + NP_SLOTS) return;
    const float* src;
    ushort_t*    dst;
    int i;
    if (tid < NQ_SLOTS) { src = wq; dst = wqb; i = tid; }
    else                { src = wp; dst = wpb; i = tid - NQ_SLOTS; }
    int lane = i & 31;
    int kt   = (i >> 5) & 15;
    int ct   = i >> 9;
    int c    = ct * 16 + (lane & 15);
    int kb   = kt * 32 + 16 * (lane >> 4);
    const float* s = src + (size_t)c * C_DIM + kb;
    ushort_t tmp[16];
    #pragma unroll
    for (int e = 0; e < 16; ++e) tmp[e] = f2bf(s[e]);
    uint4* d4 = (uint4*)(dst + (size_t)i * 16);
    d4[0] = ((uint4*)tmp)[0];
    d4[1] = ((uint4*)tmp)[1];
}

// --------------------------- fused window attention -------------------------
__global__ __launch_bounds__(512, 1)
void win_attn_kernel(const float* __restrict__ x,
                     const ushort_t* __restrict__ wqb,  // packed fragments
                     const ushort_t* __restrict__ wpb,  // packed fragments
                     const float* __restrict__ b_proj,
                     float* __restrict__ out) {
    extern __shared__ unsigned char smem_raw[];
    ushort_t* xs  = (ushort_t*)smem_raw;          // [64][520]   (region 0)
    ushort_t* qs  = xs  + R0_ELEMS;               // [64][520]   (region 1)
    ushort_t* ks  = qs  + R1_ELEMS;               // [64][520]   (region 2)
    ushort_t* vst = ks  + R2_ELEMS;               // [8][64][72] (region 3)
    float*    ls  = (float*)(vst + R3_ELEMS);     // [8][64]
    ushort_t* ps  = xs;   // overlay region 0 after stage B : [8][64][72]
    ushort_t* os  = qs;   // overlay region 1 after softmax : [64][520]

    const int b    = blockIdx.x >> 6;
    const int wloc = blockIdx.x & 63;
    const int wave = threadIdx.x >> 5;   // 0..15
    const int lane = threadIdx.x & 31;
    const int ln   = lane & 15;
    const int hl   = lane >> 4;

    // ---- stage A: load rolled x window, fp32 -> bf16 pairs, b64 LDS stores --
    {
        for (int i = threadIdx.x; i < WTOK * (C_DIM / 4); i += blockDim.x) {
            int t   = i >> 7;            // /128 float4 per row
            int p4  = i & 127;
            int seq = (wloc * WTOK + t + 32) & (SEQ - 1);     // roll -32
            float4 v = ((const float4*)(x + ((size_t)b * SEQ + seq) * C_DIM))[p4];
            unsigned int lo = ((unsigned int)f2bf(v.y) << 16) | f2bf(v.x);
            unsigned int hi = ((unsigned int)f2bf(v.w) << 16) | f2bf(v.z);
            uint2 pk; pk.x = lo; pk.y = hi;
            *(uint2*)(xs + t * LDSP + p4 * 4) = pk;
        }
    }
    __syncthreads();

    // ---- stage B: QKV gemm 64x1536, K=512.  ct -> kt -> rt so each B
    //      fragment is fetched once and reused by 4 row accumulators. ----
    for (int j = 0; j < 6; ++j) {
        int ct = wave * 6 + j;                         // 0..95
        int c  = ct * 16 + ln;
        const ushort_t* bptr = wqb + (size_t)ct * 8192 + lane * 16;
        v8f acc[4] = {};
        #pragma unroll
        for (int kt = 0; kt < 16; ++kt) {
            v16bf bb = frag_b16(bptr + kt * 512);      // 1KB/wave burst
            #pragma unroll
            for (int rt = 0; rt < 4; ++rt) {
                v16bf a = frag_a(xs + (rt * 16 + ln) * LDSP + kt * 32 + 16 * hl);
                acc[rt] = wmma_bf16(a, bb, acc[rt]);
            }
        }
        int m  = c >> 9;        // 0=q 1=k 2=v (uniform over wave)
        int cc = c & 511;       // h*64 + d
        #pragma unroll
        for (int rt = 0; rt < 4; ++rt) {
            #pragma unroll
            for (int r = 0; r < 8; ++r) {
                int t = rt * 16 + r + 8 * hl;
                float v = acc[rt][r];
                if (m == 0)      qs[t * LDSP + cc] = f2bf(v * SCALE_Q);
                else if (m == 1) ks[t * LDSP + cc] = f2bf(v);
                else {
                    int h = cc >> 6, d = cc & 63;
                    vst[h * (HDIM * PT) + d * PT + t] = f2bf(v);  // [h][d][t]
                }
            }
        }
    }
    __syncthreads();

    // ---- stage C: attention. 2 waves per head; each wave owns 2 strips of
    //      16 rows of the 64x64 score matrix. ----
    const int h    = wave >> 1;
    const int half = wave & 1;
    v8f S[2][4];
    #pragma unroll
    for (int s = 0; s < 2; ++s) {
        int rt = half * 2 + s;
        const ushort_t* abase = qs + (rt * 16 + ln) * LDSP + h * HDIM + 16 * hl;
        v16bf a0 = frag_a(abase);
        v16bf a1 = frag_a(abase + 32);
        #pragma unroll
        for (int ct = 0; ct < 4; ++ct) {
            const ushort_t* bbase = ks + (ct * 16 + ln) * LDSP + h * HDIM + 16 * hl;
            v8f acc = {};
            acc = wmma_bf16(a0, frag_b16(bbase), acc);
            acc = wmma_bf16(a1, frag_b16(bbase + 32), acc);
            S[s][ct] = acc;
        }
    }
    // softmax: row = rt*16 + r + 8*hl lives in one 16-lane group
    #pragma unroll
    for (int s = 0; s < 2; ++s) {
        int rt = half * 2 + s;
        #pragma unroll
        for (int r = 0; r < 8; ++r) {
            float mx = -3.402823466e38f;
            #pragma unroll
            for (int ct = 0; ct < 4; ++ct) mx = fmaxf(mx, S[s][ct][r]);
            #pragma unroll
            for (int off = 8; off >= 1; off >>= 1)
                mx = fmaxf(mx, __shfl_xor(mx, off, 32));
            float p[4]; float sum = 0.f;
            #pragma unroll
            for (int ct = 0; ct < 4; ++ct) { p[ct] = __expf(S[s][ct][r] - mx); sum += p[ct]; }
            #pragma unroll
            for (int off = 8; off >= 1; off >>= 1)
                sum += __shfl_xor(sum, off, 32);
            int row = rt * 16 + r + 8 * hl;
            if (ln == 0) ls[h * WTOK + row] = sum;
            #pragma unroll
            for (int ct = 0; ct < 4; ++ct)
                ps[h * (WTOK * PT) + row * PT + ct * 16 + ln] = f2bf(p[ct]);
        }
    }
    __syncthreads();   // all waves done with qs/xs reads; ps complete

    // ---- O = P * V (per head), normalize by row sum, bf16 to os ----
    #pragma unroll
    for (int s = 0; s < 2; ++s) {
        int rt = half * 2 + s;
        const ushort_t* abase =
            ps + h * (WTOK * PT) + (rt * 16 + ln) * PT + 16 * hl;
        v16bf a0 = frag_a(abase);
        v16bf a1 = frag_a(abase + 32);
        float inv[8];
        #pragma unroll
        for (int r = 0; r < 8; ++r)
            inv[r] = 1.0f / ls[h * WTOK + rt * 16 + r + 8 * hl];
        #pragma unroll
        for (int ct = 0; ct < 4; ++ct) {               // output d tiles
            const ushort_t* bbase =
                vst + h * (HDIM * PT) + (ct * 16 + ln) * PT + 16 * hl;
            v8f acc = {};
            acc = wmma_bf16(a0, frag_b16(bbase), acc);
            acc = wmma_bf16(a1, frag_b16(bbase + 32), acc);
            #pragma unroll
            for (int r = 0; r < 8; ++r) {
                int t = rt * 16 + r + 8 * hl;
                os[t * LDSP + h * HDIM + ct * 16 + ln] = f2bf(acc[r] * inv[r]);
            }
        }
    }
    __syncthreads();

    // ---- stage D: proj gemm 64x512, K=512, +bias, fp32 out (inverse roll).
    //      2 col-tiles per wave; B fragment shared across 4 row tiles. ----
    for (int j = 0; j < 2; ++j) {
        int ct = wave * 2 + j;                         // 0..31
        int n  = ct * 16 + ln;
        const ushort_t* bptr = wpb + (size_t)ct * 8192 + lane * 16;
        v8f acc[4] = {};
        #pragma unroll
        for (int kt = 0; kt < 16; ++kt) {
            v16bf bb = frag_b16(bptr + kt * 512);
            #pragma unroll
            for (int rt = 0; rt < 4; ++rt) {
                v16bf a = frag_a(os + (rt * 16 + ln) * LDSP + kt * 32 + 16 * hl);
                acc[rt] = wmma_bf16(a, bb, acc[rt]);
            }
        }
        float bias = b_proj[n];
        #pragma unroll
        for (int rt = 0; rt < 4; ++rt) {
            #pragma unroll
            for (int r = 0; r < 8; ++r) {
                int t    = rt * 16 + r + 8 * hl;
                int seqo = (wloc * WTOK + t + 32) & (SEQ - 1);  // roll +32
                out[((size_t)b * SEQ + seqo) * C_DIM + n] = acc[rt][r] + bias;
            }
        }
    }
}

// ---------------------------------------------------------------------------
extern "C" void kernel_launch(void* const* d_in, const int* in_sizes, int n_in,
                              void* d_out, int out_size, void* d_ws, size_t ws_size,
                              hipStream_t stream) {
    const float* x      = (const float*)d_in[0];
    const float* w_qkv  = (const float*)d_in[1];
    const float* w_proj = (const float*)d_in[2];
    const float* b_proj = (const float*)d_in[3];
    float* out = (float*)d_out;

    ushort_t* wqb = (ushort_t*)d_ws;                      // NQ_SLOTS*16 bf16
    ushort_t* wpb = wqb + (size_t)NQ_SLOTS * 16;          // NP_SLOTS*16 bf16

    pack_weights_kernel<<<(NQ_SLOTS + NP_SLOTS + 255) / 256, 256, 0, stream>>>(
        w_qkv, w_proj, wqb, wpb);

    // LDS: 36864 + 33280 + 33280 + 36864 bf16 + 512 floats = 282,624 B
    size_t smem = (size_t)(R0_ELEMS + R1_ELEMS + R2_ELEMS + R3_ELEMS)
                      * sizeof(ushort_t)
                + NHEAD * WTOK * sizeof(float);
    win_attn_kernel<<<dim3(8 * NWINB), dim3(512), smem, stream>>>(
        x, wqb, wpb, b_proj, out);
}
